// Attention_19318762897638
// MI455X (gfx1250) — matrix operand order repaired and checked by reading, not compiled, not run on hardware
//
#include <hip/hip_runtime.h>
#include <hip/hip_bf16.h>

typedef __attribute__((ext_vector_type(16))) _Float16 v16h;
typedef __attribute__((ext_vector_type(8)))  _Float16 v8h;
typedef __attribute__((ext_vector_type(8)))  float    v8f;

#define B_   2
#define S_   2048
#define NX_  1024
#define NH_  16
#define HD_  64
#define N3X_ 3072

// ---------- helpers ----------
static __device__ __forceinline__ v8f vsplat8(float x) {
  v8f v;
#pragma unroll
  for (int i = 0; i < 8; ++i) v[i] = x;
  return v;
}

// max-reduce + broadcast across the 16 lanes of a half-row.
// Prefer v_permlane16_b32 (VALU pipe, keeps the LDS pipe free for P staging);
// fall back to __shfl_xor (ds_bpermute) if the builtin is unavailable.
static __device__ __forceinline__ float bcast_max16(float x) {
#if __has_builtin(__builtin_amdgcn_permlane16)
  union { float f; int i; } u, v;
  u.f = x;
  v.i = __builtin_amdgcn_permlane16(u.i, u.i, 0x67452301u, 0xEFCDAB89u, false, false);
  u.f = fmaxf(u.f, v.f);  // xor 1
  v.i = __builtin_amdgcn_permlane16(u.i, u.i, 0x54761032u, 0xDCFE98BAu, false, false);
  u.f = fmaxf(u.f, v.f);  // xor 2
  v.i = __builtin_amdgcn_permlane16(u.i, u.i, 0x32107654u, 0xBA98FEDCu, false, false);
  u.f = fmaxf(u.f, v.f);  // xor 4
  v.i = __builtin_amdgcn_permlane16(u.i, u.i, 0xFEDCBA98u, 0x76543210u, false, false);
  u.f = fmaxf(u.f, v.f);  // xor 8
  return u.f;
#else
#pragma unroll
  for (int off = 1; off < 16; off <<= 1) x = fmaxf(x, __shfl_xor(x, off));
  return x;
#endif
}

// A-fragment (16x32 f16, row-major source, stride ld):
// lane l: row = l&15, K elems {kb..kb+7} and {16+kb..16+kb+7}, kb = 8*(l>>4)
static __device__ __forceinline__ v16h frag_a_load(const _Float16* base, int row0,
                                                   int ld, int kbase, int lane) {
  const int m  = lane & 15;
  const int kb = (lane >> 4) << 3;
  const _Float16* p = base + (size_t)(row0 + m) * ld + kbase + kb;
  v8h lo = *reinterpret_cast<const v8h*>(p);
  v8h hi = *reinterpret_cast<const v8h*>(p + 16);
  v16h r;
#pragma unroll
  for (int i = 0; i < 8; ++i) { r[i] = lo[i]; r[i + 8] = hi[i]; }
  return r;
}

// B-fragment (32x16 f16) loaded from B-transposed storage Bt[n][k] (stride ld):
// lane l: col = l&15, K elems {kb..kb+7} and {16+kb..16+kb+7}, kb = 8*(l>>4)
static __device__ __forceinline__ v16h frag_b_load(const _Float16* baseT, int col0,
                                                   int ld, int kbase, int lane) {
  const int n  = lane & 15;
  const int kb = (lane >> 4) << 3;
  const _Float16* p = baseT + (size_t)(col0 + n) * ld + kbase + kb;
  v8h lo = *reinterpret_cast<const v8h*>(p);
  v8h hi = *reinterpret_cast<const v8h*>(p + 16);
  v16h r;
#pragma unroll
  for (int i = 0; i < 8; ++i) { r[i] = lo[i]; r[i + 8] = hi[i]; }
  return r;
}

static __device__ __forceinline__ v8f wmma16(v16h a, v16h b, v8f c) {
  return __builtin_amdgcn_wmma_f32_16x16x32_f16(false, a, false, b, (short)0, c,
                                                false, false);
}

// ---------- conversion kernels ----------
__global__ void cvt_f32_f16_k(const float* __restrict__ src,
                              _Float16* __restrict__ dst, int n) {
  int i = blockIdx.x * blockDim.x + threadIdx.x;
  if (i < n) dst[i] = (_Float16)src[i];
}

// src [R][C] f32 -> dst [C][R] f16
__global__ void transpose_f32_f16_k(const float* __restrict__ src,
                                    _Float16* __restrict__ dst, int R, int C) {
  int i = blockIdx.x * blockDim.x + threadIdx.x;
  if (i < R * C) {
    int c = i / R;
    int r = i - c * R;
    dst[i] = (_Float16)src[(size_t)r * C + c];
  }
}

// ---------- WMMA GEMM: C[M][N] = A[M][K] * Bt[N][K]^T + bias ----------
// EPI==0: QKV epilogue -> scatter f16 into Q[b,h,s,d], K[b,h,s,d], Vt[b,h,d,s]
// EPI==1: fp32 output (final projection)
template <int EPI>
__global__ __launch_bounds__(128) void gemm_wmma_k(
    const _Float16* __restrict__ A, const _Float16* __restrict__ Bt,
    const float* __restrict__ bias, _Float16* __restrict__ Qh,
    _Float16* __restrict__ Kh, _Float16* __restrict__ Vt,
    float* __restrict__ Out, int N, int Kd) {
  const int lane = threadIdx.x & 31;
  const int wave = threadIdx.x >> 5;
  const int row0 = blockIdx.x * 64;
  const int col0 = blockIdx.y * 256 + wave * 64;

  v8f acc[4][4];
#pragma unroll
  for (int i = 0; i < 4; ++i)
#pragma unroll
    for (int j = 0; j < 4; ++j) acc[i][j] = vsplat8(0.0f);

  for (int kk = 0; kk < Kd; kk += 32) {
    v16h af[4], bfr[4];
#pragma unroll
    for (int mi = 0; mi < 4; ++mi) af[mi] = frag_a_load(A, row0 + mi * 16, Kd, kk, lane);
#pragma unroll
    for (int ni = 0; ni < 4; ++ni) bfr[ni] = frag_b_load(Bt, col0 + ni * 16, Kd, kk, lane);
    if (kk + 32 < Kd) {  // pull next K-slice toward WGP$/L2 (global_prefetch_b8)
      __builtin_prefetch(A + (size_t)(row0 + (lane & 15)) * Kd + kk + 32, 0, 3);
      __builtin_prefetch(Bt + (size_t)(col0 + (lane & 15)) * Kd + kk + 32, 0, 3);
    }
#pragma unroll
    for (int mi = 0; mi < 4; ++mi)
#pragma unroll
      for (int ni = 0; ni < 4; ++ni)
        acc[mi][ni] = wmma16(af[mi], bfr[ni], acc[mi][ni]);
  }

  const int rsub = (lane >> 4) << 3;  // row offset within 16-tile
  const int nsub = lane & 15;        // col within 16-tile
#pragma unroll
  for (int mi = 0; mi < 4; ++mi)
#pragma unroll
    for (int ni = 0; ni < 4; ++ni) {
      const int gc = col0 + ni * 16 + nsub;
      const float bv = bias[gc];
#pragma unroll
      for (int r = 0; r < 8; ++r) {
        const int gr = row0 + mi * 16 + rsub + r;
        const float v = acc[mi][ni][r] + bv;
        if (EPI == 0) {
          const int sec = gc >> 10;          // 0=Q 1=K 2=V
          const int c   = gc & (NX_ - 1);
          const int h   = c >> 6;
          const int d   = c & 63;
          const int b   = gr >> 11;          // / S_
          const int s   = gr & (S_ - 1);
          const size_t bh = (size_t)b * NH_ + h;
          if (sec == 0)      Qh[(bh * S_ + s) * HD_ + d] = (_Float16)v;
          else if (sec == 1) Kh[(bh * S_ + s) * HD_ + d] = (_Float16)v;
          else               Vt[(bh * HD_ + d) * S_ + s] = (_Float16)v;
        } else {
          Out[(size_t)gr * N + gc] = v;
        }
      }
    }
}

// ---------- flash attention: one wave = one 64-row q-tile of one (b,h) ----------
__global__ __launch_bounds__(128) void attn_k(const _Float16* __restrict__ Qh,
                                              const _Float16* __restrict__ Kh,
                                              const _Float16* __restrict__ Vt,
                                              _Float16* __restrict__ AO) {
  __shared__ _Float16 ldsP[4 * 64 * 32];  // per-wave 64x32 P tile
  const int lane = threadIdx.x & 31;
  const int wave = threadIdx.x >> 5;
  const int w  = blockIdx.x * 4 + wave;   // 0..1023
  const int qt = w & 31;                  // S_/64 = 32 q-tiles
  const int bh = w >> 5;                  // 0..31 (b*NH+h)
  const int b  = bh >> 4;
  const int h  = bh & 15;
  const int q0 = qt * 64;

  const _Float16* Qp = Qh + (size_t)bh * S_ * HD_;
  const _Float16* Kp = Kh + (size_t)bh * S_ * HD_;
  const _Float16* Vp = Vt + (size_t)bh * HD_ * S_;
  _Float16* lp = &ldsP[wave * 64 * 32];

  // constant all-ones B-fragment: rowsum(P) = P x ones via the matrix unit
  v16h fones;
#pragma unroll
  for (int i = 0; i < 16; ++i) fones[i] = (_Float16)1.0f;

  // Q fragments, kept resident, pre-scaled by 1/sqrt(HD)=0.125 (exact in f16)
  v16h aq[4][2];
#pragma unroll
  for (int mi = 0; mi < 4; ++mi)
#pragma unroll
    for (int dk = 0; dk < 2; ++dk) {
      aq[mi][dk] = frag_a_load(Qp, q0 + mi * 16, HD_, dk * 32, lane);
#pragma unroll
      for (int e = 0; e < 16; ++e) aq[mi][dk][e] *= (_Float16)0.125f;
    }

  v8f O[4][4], mrun[4], lrun[4];
#pragma unroll
  for (int mi = 0; mi < 4; ++mi) {
    mrun[mi] = vsplat8(-1e30f);
    lrun[mi] = vsplat8(0.0f);
#pragma unroll
    for (int ni = 0; ni < 4; ++ni) O[mi][ni] = vsplat8(0.0f);
  }

  const int rsub = (lane >> 4) << 3;
  const int nsub = lane & 15;

  // one 32-key block; `masked` is a compile-time constant at each call site
  auto kblock = [&](const int k0, const bool masked) {
    // ---- S = Q K^T (B-operand of QK^T is K itself since Bt = (K^T)^T) ----
    v16h bk[2][2];
#pragma unroll
    for (int ni = 0; ni < 2; ++ni)
#pragma unroll
      for (int dk = 0; dk < 2; ++dk)
        bk[ni][dk] = frag_b_load(Kp, k0 + ni * 16, HD_, dk * 32, lane);

    v8f sacc[4][2];
#pragma unroll
    for (int mi = 0; mi < 4; ++mi)
#pragma unroll
      for (int ni = 0; ni < 2; ++ni) {
        sacc[mi][ni] = vsplat8(0.0f);
        sacc[mi][ni] = wmma16(aq[mi][0], bk[ni][0], sacc[mi][ni]);
        sacc[mi][ni] = wmma16(aq[mi][1], bk[ni][1], sacc[mi][ni]);
      }

    // ---- (optional) causal mask + online softmax (max via permlane) ----
    v8f corr[4];
#pragma unroll
    for (int mi = 0; mi < 4; ++mi) {
      if (masked) {
#pragma unroll
        for (int ni = 0; ni < 2; ++ni) {
          const int kc = k0 + ni * 16 + nsub;
#pragma unroll
          for (int r = 0; r < 8; ++r) {
            const int qr = q0 + mi * 16 + rsub + r;
            sacc[mi][ni][r] = (kc <= qr) ? sacc[mi][ni][r] : -1e30f;
          }
        }
      }
      v8f mt;
#pragma unroll
      for (int r = 0; r < 8; ++r)
        mt[r] = bcast_max16(fmaxf(sacc[mi][0][r], sacc[mi][1][r]));
#pragma unroll
      for (int r = 0; r < 8; ++r) {
        const float mnew = fmaxf(mrun[mi][r], mt[r]);
        corr[mi][r] = __expf(mrun[mi][r] - mnew);
        mrun[mi][r] = mnew;
      }
#pragma unroll
      for (int ni = 0; ni < 2; ++ni)
#pragma unroll
        for (int r = 0; r < 8; ++r)
          sacc[mi][ni][r] = __expf(sacc[mi][ni][r] - mrun[mi][r]);
#pragma unroll
      for (int ni = 0; ni < 4; ++ni)
#pragma unroll
        for (int r = 0; r < 8; ++r) O[mi][ni][r] *= corr[mi][r];
      // stage P (C-layout -> LDS, row-major 64x32 f16)
#pragma unroll
      for (int ni = 0; ni < 2; ++ni)
#pragma unroll
        for (int r = 0; r < 8; ++r)
          lp[(mi * 16 + rsub + r) * 32 + ni * 16 + nsub] = (_Float16)sacc[mi][ni][r];
    }

    // ---- O += P V and lrun += rowsum(P) (both on the matrix unit) ----
    v16h bv[4];
#pragma unroll
    for (int ni = 0; ni < 4; ++ni) bv[ni] = frag_b_load(Vp, ni * 16, S_, k0, lane);
#pragma unroll
    for (int mi = 0; mi < 4; ++mi) {
      const int m  = lane & 15;
      const int kb = (lane >> 4) << 3;
      const _Float16* pp = lp + (mi * 16 + m) * 32;
      v8h lo = *reinterpret_cast<const v8h*>(pp + kb);
      v8h hi = *reinterpret_cast<const v8h*>(pp + 16 + kb);
      v16h ap;
#pragma unroll
      for (int i = 0; i < 8; ++i) { ap[i] = lo[i]; ap[i + 8] = hi[i]; }
      // rowsum(P): P x ones -> every lane holds its 8 row-sums
      const v8f rs = wmma16(ap, fones, vsplat8(0.0f));
#pragma unroll
      for (int r = 0; r < 8; ++r)
        lrun[mi][r] = lrun[mi][r] * corr[mi][r] + rs[r];
#pragma unroll
      for (int ni = 0; ni < 4; ++ni) O[mi][ni] = wmma16(ap, bv[ni], O[mi][ni]);
    }
  };

  // causal structure: blocks 0..2*qt-1 are fully below the diagonal (no mask
  // needed: max key k0+31 <= q0 <= min row); only the last two 32-key blocks
  // straddle the diagonal and need per-element masking.
  const int nfull = qt * 2;
  for (int kb_i = 0; kb_i < nfull; ++kb_i) kblock(kb_i * 32, false);
  kblock(nfull * 32, true);
  kblock(nfull * 32 + 32, true);

  // ---- epilogue: normalize, merge heads, store f16 ----
#pragma unroll
  for (int mi = 0; mi < 4; ++mi)
#pragma unroll
    for (int ni = 0; ni < 4; ++ni)
#pragma unroll
      for (int r = 0; r < 8; ++r) {
        const int sloc = q0 + mi * 16 + rsub + r;
        const int d = ni * 16 + nsub;
        const float v = O[mi][ni][r] / lrun[mi][r];
        AO[((size_t)b * S_ + sloc) * NX_ + h * HD_ + d] = (_Float16)v;
      }
}

// ---------- host launch ----------
extern "C" void kernel_launch(void* const* d_in, const int* in_sizes, int n_in,
                              void* d_out, int out_size, void* d_ws, size_t ws_size,
                              hipStream_t stream) {
  const float* x    = (const float*)d_in[0];
  const float* wqkv = (const float*)d_in[1];
  const float* bqkv = (const float*)d_in[2];
  const float* wprj = (const float*)d_in[3];
  const float* bprj = (const float*)d_in[4];
  float* out = (float*)d_out;

  char* ws = (char*)d_ws;
  size_t off = 0;
  auto take = [&](size_t bytes) {
    void* p = ws + off;
    off += (bytes + 255) & ~(size_t)255;
    return p;
  };
  _Float16* Xh  = (_Float16*)take((size_t)B_ * S_ * NX_ * 2);
  _Float16* Wqt = (_Float16*)take((size_t)N3X_ * NX_ * 2);
  _Float16* Wpt = (_Float16*)take((size_t)NX_ * NX_ * 2);
  _Float16* Qh  = (_Float16*)take((size_t)B_ * NH_ * S_ * HD_ * 2);
  _Float16* Kh  = (_Float16*)take((size_t)B_ * NH_ * S_ * HD_ * 2);
  _Float16* Vt  = (_Float16*)take((size_t)B_ * NH_ * S_ * HD_ * 2);
  _Float16* AO  = (_Float16*)take((size_t)B_ * S_ * NX_ * 2);

  const int nX = B_ * S_ * NX_;
  cvt_f32_f16_k<<<(nX + 255) / 256, 256, 0, stream>>>(x, Xh, nX);
  transpose_f32_f16_k<<<(NX_ * N3X_ + 255) / 256, 256, 0, stream>>>(wqkv, Wqt, NX_, N3X_);
  transpose_f32_f16_k<<<(NX_ * NX_ + 255) / 256, 256, 0, stream>>>(wprj, Wpt, NX_, NX_);

  // QKV: M=4096, N=3072, K=1024 ; block = 4 waves -> 64 rows x 256 cols
  gemm_wmma_k<0><<<dim3(64, 12), 128, 0, stream>>>(Xh, Wqt, bqkv, Qh, Kh, Vt,
                                                   nullptr, N3X_, NX_);
  // attention: 1024 q-tile waves, 4 per block
  attn_k<<<256, 128, 0, stream>>>(Qh, Kh, Vt, AO);
  // projection: M=4096, N=1024, K=1024, fp32 out
  gemm_wmma_k<1><<<dim3(64, 4), 128, 0, stream>>>(AO, Wpt, bprj, nullptr, nullptr,
                                                  nullptr, out, NX_, NX_);
}
